// DeformableTransformer_23974507446522
// MI455X (gfx1250) — matrix operand order repaired and checked by reading, not compiled, not run on hardware
//
#include <hip/hip_runtime.h>
#include <hip/hip_bf16.h>
#include <math.h>

typedef __attribute__((ext_vector_type(16))) _Float16 v16h;
typedef __attribute__((ext_vector_type(8)))  _Float16 v8h;
typedef __attribute__((ext_vector_type(4)))  _Float16 v4h;
typedef __attribute__((ext_vector_type(8)))  float    v8f;

#define D_MODEL 384
#define NHEAD   8
#define HEAD_DIM 48
#define NPTS    4
#define LENC    6
#define LDEC    6
#define DFFN    1536
#define BS      2
#define SD      12
#define SH      12
#define SW      12
#define NQ      300
#define SVOX    (SD*SH*SW)      /* 1728 */
#define MB_ENC  (BS*SVOX)       /* 3456 rows, encoder token count */
#define MD_DEC  (BS*NQ)         /* 600 rows, decoder token count */

// ---------------------------------------------------------------------------
// f16 fragment loads (operands pre-converted to f16 in workspace).
// A fragment (16x32): lane holds row m, K elems {d..d+7, d+16..d+23}, d=(lane&16)?8:0
//   -> two 16-byte contiguous runs -> 2x global_load_b128, zero conversion.
// B fragment (32x16): lane holds col n, K elems {d..d+7, d+16..d+23}, d=(lane&16)?8:0
//   -> two 16-byte contiguous runs -> 2x global_load_b128.
// ---------------------------------------------------------------------------
__device__ __forceinline__ v16h loadA16(const _Float16* __restrict__ p) {
    v8h lo = *(const v8h*)(p);
    v8h hi = *(const v8h*)(p + 16);
    v16h r;
    #pragma unroll
    for (int i = 0; i < 8; ++i) { r[i] = lo[i]; r[i + 8] = hi[i]; }
    return r;
}

__device__ __forceinline__ v16h loadB16(const _Float16* __restrict__ p) {
    v8h lo = *(const v8h*)(p);
    v8h hi = *(const v8h*)(p + 16);
    v16h r;
    #pragma unroll
    for (int i = 0; i < 8; ++i) { r[i] = lo[i]; r[i + 8] = hi[i]; }
    return r;
}

// ---------------------------------------------------------------------------
// GEMM: Y(M,N) = act( Ah(M,K) @ Wh(N,K)^T + bias ), f16 operands, f32 accum.
// One wave -> 16 (M) x (16*NT) (N) tile: one A fragment feeds NT WMMAs.
// Requires N % (16*NT) == 0, K % 32 == 0. M guarded. EXEC all-ones at WMMA.
// ---------------------------------------------------------------------------
template<int NT>
__global__ __launch_bounds__(256) void gemm_wmma_f16(
    const _Float16* __restrict__ A, const _Float16* __restrict__ W,
    const float* __restrict__ bias, float* __restrict__ Y,
    int M, int N, int K, int relu)
{
    const int wid  = (blockIdx.x * blockDim.x + threadIdx.x) >> 5;
    const int lane = threadIdx.x & 31;
    const int tilesM = (M + 15) >> 4;
    const int tilesN = N / (16 * NT);
    if (wid >= tilesM * tilesN) return;

    const int tm = wid / tilesN;
    const int tn = wid % tilesN;
    const int m0 = tm << 4;
    const int n0 = tn * (16 * NT);

    const int mA   = m0 + (lane & 15);
    const bool mOK = (mA < M);
    const int dA   = (lane & 16) ? 8  : 0;
    const int dB   = (lane & 16) ? 8  : 0;

    v8f acc[NT];
    #pragma unroll
    for (int t = 0; t < NT; ++t) acc[t] = (v8f){};

    const _Float16* Arow = A + (size_t)mA * K + dA;
    const _Float16* Brow[NT];
    #pragma unroll
    for (int t = 0; t < NT; ++t)
        Brow[t] = W + (size_t)(n0 + t*16 + (lane & 15)) * K + dB;

    for (int k = 0; k < K; k += 32) {
        v16h af;
        if (mOK) {
            af = loadA16(Arow + k);
        } else {
            #pragma unroll
            for (int i = 0; i < 16; ++i) af[i] = (_Float16)0.0f;
        }
        #pragma unroll
        for (int t = 0; t < NT; ++t) {
            v16h bf = loadB16(Brow[t] + k);
            acc[t] = __builtin_amdgcn_wmma_f32_16x16x32_f16(false, af, false, bf,
                                                            (short)0, acc[t], false, false);
        }
    }

    const int mS = m0 + ((lane & 16) ? 8 : 0);
    #pragma unroll
    for (int t = 0; t < NT; ++t) {
        const int n = n0 + t*16 + (lane & 15);
        const float bb = bias[n];
        #pragma unroll
        for (int j = 0; j < 8; ++j) {
            const int m = mS + j;
            if (m < M) {
                float v = acc[t][j] + bb;
                if (relu) v = fmaxf(v, 0.0f);
                Y[(size_t)m * N + n] = v;
            }
        }
    }
}

// 4-wide f32 -> f16 conversion (all tensor sizes here are multiples of 4)
__global__ void k_cvt_f16(const float* __restrict__ in, _Float16* __restrict__ out, int n4) {
    int i = blockIdx.x * blockDim.x + threadIdx.x;
    if (i < n4) {
        float4 f = ((const float4*)in)[i];
        v4h h;
        h[0] = (_Float16)f.x; h[1] = (_Float16)f.y;
        h[2] = (_Float16)f.z; h[3] = (_Float16)f.w;
        ((v4h*)out)[i] = h;
    }
}

// ---------------------------------------------------------------------------
// LayerNorm with fused residual: out = LN(x + x2) * g + b   (C == 384)
// ---------------------------------------------------------------------------
__global__ __launch_bounds__(128) void ln_residual(
    const float* __restrict__ x, const float* __restrict__ x2,
    const float* __restrict__ g, const float* __restrict__ b,
    float* __restrict__ out)
{
    const int row = blockIdx.x;
    const int t   = threadIdx.x;
    const float* px = x  + (size_t)row * D_MODEL;
    const float* p2 = x2 + (size_t)row * D_MODEL;
    float v[3]; float s = 0.0f;
    #pragma unroll
    for (int i = 0; i < 3; ++i) { v[i] = px[t + i*128] + p2[t + i*128]; s += v[i]; }

    __shared__ float red[128];
    red[t] = s; __syncthreads();
    #pragma unroll
    for (int o = 64; o > 0; o >>= 1) { if (t < o) red[t] += red[t + o]; __syncthreads(); }
    const float mean = red[0] * (1.0f / D_MODEL);
    __syncthreads();

    float s2 = 0.0f;
    #pragma unroll
    for (int i = 0; i < 3; ++i) { const float d = v[i] - mean; s2 += d * d; }
    red[t] = s2; __syncthreads();
    #pragma unroll
    for (int o = 64; o > 0; o >>= 1) { if (t < o) red[t] += red[t + o]; __syncthreads(); }
    const float rstd = rsqrtf(red[0] * (1.0f / D_MODEL) + 1e-5f);

    #pragma unroll
    for (int i = 0; i < 3; ++i) {
        const int c = t + i*128;
        out[(size_t)row * D_MODEL + c] = (v[i] - mean) * rstd * g[c] + b[c];
    }
}

// ---------------------------------------------------------------------------
// Small elementwise / prep kernels
// ---------------------------------------------------------------------------
__global__ void k_add2(const float* __restrict__ a, const float* __restrict__ b,
                       float* __restrict__ o, int n) {
    int i = blockIdx.x * blockDim.x + threadIdx.x;
    if (i < n) o[i] = a[i] + b[i];
}

__global__ void k_copy(const float* __restrict__ a, float* __restrict__ o, int n) {
    int i = blockIdx.x * blockDim.x + threadIdx.x;
    if (i < n) o[i] = a[i];
}

__global__ void k_prep_xpos(const float* __restrict__ srcs,
                            const float* __restrict__ pose,
                            const float* __restrict__ lvl,
                            float* __restrict__ x, float* __restrict__ pos) {
    int idx = blockIdx.x * blockDim.x + threadIdx.x;
    if (idx >= BS * SVOX * D_MODEL) return;
    int c = idx % D_MODEL;
    int s = (idx / D_MODEL) % SVOX;
    int b = idx / (D_MODEL * SVOX);
    size_t src = ((size_t)b * D_MODEL + c) * SVOX + s;
    x[idx]   = srcs[src];
    pos[idx] = pose[src] + lvl[c];
}

__global__ void k_valid(const unsigned char* __restrict__ mask, float* __restrict__ valid) {
    int b = threadIdx.x;
    if (b >= BS) return;
    int cw = 0, ch = 0, cd = 0;
    for (int i = 0; i < 12; ++i) {
        cw += mask[b*SVOX + i]       ? 0 : 1;   // m[b,0,0,x]
        ch += mask[b*SVOX + i*SW]    ? 0 : 1;   // m[b,0,y,0]
        cd += mask[b*SVOX + i*SH*SW] ? 0 : 1;   // m[b,z,0,0]
    }
    valid[b*3+0] = cw / (float)SW;
    valid[b*3+1] = ch / (float)SH;
    valid[b*3+2] = cd / (float)SD;
}

__global__ void k_enc_ref(const float* __restrict__ valid, float* __restrict__ ref) {
    int idx = blockIdx.x * blockDim.x + threadIdx.x;
    if (idx >= BS * SVOX) return;
    int s = idx % SVOX;
    int b = idx / SVOX;
    int z = s / (SH*SW);
    int y = (s / SW) % SH;
    int x = s % SW;
    ref[(size_t)idx*3+0] = ((x + 0.5f) / SW) * valid[b*3+0];
    ref[(size_t)idx*3+1] = ((y + 0.5f) / SH) * valid[b*3+1];
    ref[(size_t)idx*3+2] = ((z + 0.5f) / SD) * valid[b*3+2];
}

__global__ void k_mask_zero(float* __restrict__ val, const unsigned char* __restrict__ mask) {
    int idx = blockIdx.x * blockDim.x + threadIdx.x;
    if (idx >= MB_ENC * D_MODEL) return;
    int row = idx / D_MODEL;               // (b*S + s)
    if (mask[row]) val[idx] = 0.0f;
}

__global__ void k_init_query(const float* __restrict__ qemb,
                             float* __restrict__ qe, float* __restrict__ y) {
    int idx = blockIdx.x * blockDim.x + threadIdx.x;
    if (idx >= MD_DEC * D_MODEL) return;
    int c = idx % D_MODEL;
    int q = (idx / D_MODEL) % NQ;
    qe[idx] = qemb[(size_t)q * 2*D_MODEL + c];
    y[idx]  = qemb[(size_t)q * 2*D_MODEL + D_MODEL + c];
}

__global__ void k_dec_ref(const float* __restrict__ qe, const float* __restrict__ rw,
                          const float* __restrict__ rb, const float* __restrict__ valid,
                          float* __restrict__ dref) {
    int idx = blockIdx.x * blockDim.x + threadIdx.x;
    if (idx >= MD_DEC * 3) return;
    int j = idx % 3;
    int r = idx / 3;          // b*NQ + q
    int b = r / NQ;
    const float* x = qe + (size_t)r * D_MODEL;
    const float* w = rw + (size_t)j * D_MODEL;
    float acc = rb[j];
    for (int k = 0; k < D_MODEL; ++k) acc += x[k] * w[k];
    dref[idx] = (1.0f / (1.0f + expf(-acc))) * valid[b*3+j];
}

// ---------------------------------------------------------------------------
// MSDA trilinear sampling + attention-weight softmax + weighted sum.
// ---------------------------------------------------------------------------
__global__ void k_msda_sample(const float* __restrict__ value,
                              const float* __restrict__ off,
                              const float* __restrict__ awl,
                              const float* __restrict__ ref,
                              float* __restrict__ out, int Lq)
{
    int idx = blockIdx.x * blockDim.x + threadIdx.x;
    if (idx >= BS * Lq * D_MODEL) return;
    int c = idx % HEAD_DIM;
    int h = (idx / HEAD_DIM) % NHEAD;
    int q = (idx / D_MODEL) % Lq;
    int b = idx / (D_MODEL * Lq);

    const size_t bq = (size_t)b * Lq + q;
    const float rx = ref[bq*3+0], ry = ref[bq*3+1], rz = ref[bq*3+2];

    const float* ap = awl + (bq * NHEAD + h) * NPTS;
    float a0 = ap[0], a1 = ap[1], a2 = ap[2], a3 = ap[3];
    float mx = fmaxf(fmaxf(a0, a1), fmaxf(a2, a3));
    float e0 = expf(a0-mx), e1 = expf(a1-mx), e2 = expf(a2-mx), e3 = expf(a3-mx);
    float inv = 1.0f / (e0+e1+e2+e3);
    float aw[4] = {e0*inv, e1*inv, e2*inv, e3*inv};

    const float* op = off + (bq * NHEAD + h) * NPTS * 3;
    const float* vb = value + (size_t)b * SVOX * D_MODEL + h * HEAD_DIM + c;

    float acc = 0.0f;
    #pragma unroll
    for (int p = 0; p < NPTS; ++p) {
        float fxp = (rx + op[p*3+0] * (1.0f/SW)) * SW - 0.5f;
        float fyp = (ry + op[p*3+1] * (1.0f/SH)) * SH - 0.5f;
        float fzp = (rz + op[p*3+2] * (1.0f/SD)) * SD - 0.5f;
        float x0f = floorf(fxp), y0f = floorf(fyp), z0f = floorf(fzp);
        float fx = fxp - x0f, fy = fyp - y0f, fz = fzp - z0f;
        int x0 = (int)x0f, y0 = (int)y0f, z0 = (int)z0f;
        #pragma unroll
        for (int dz = 0; dz < 2; ++dz) {
            int zi = z0 + dz; float wz = dz ? fz : 1.0f - fz;
            #pragma unroll
            for (int dy = 0; dy < 2; ++dy) {
                int yi = y0 + dy; float wy = dy ? fy : 1.0f - fy;
                #pragma unroll
                for (int dx = 0; dx < 2; ++dx) {
                    int xi = x0 + dx; float wx = dx ? fx : 1.0f - fx;
                    if (xi >= 0 && xi < SW && yi >= 0 && yi < SH && zi >= 0 && zi < SD) {
                        int s = (zi*SH + yi)*SW + xi;
                        acc += aw[p] * wz * wy * wx * vb[(size_t)s * D_MODEL];
                    }
                }
            }
        }
    }
    out[idx] = acc;
}

// ---------------------------------------------------------------------------
// Decoder MHA pieces (small: L=300)
// ---------------------------------------------------------------------------
__global__ void k_attn_scores(const float* __restrict__ qk, float* __restrict__ attn) {
    int idx = blockIdx.x * blockDim.x + threadIdx.x;
    if (idx >= BS * NHEAD * NQ * NQ) return;
    int ki = idx % NQ;
    int qi = (idx / NQ) % NQ;
    int h  = (idx / (NQ*NQ)) % NHEAD;
    int b  = idx / (NQ*NQ*NHEAD);
    const float* qp = qk + ((size_t)b*NQ + qi) * (2*D_MODEL) + h*HEAD_DIM;
    const float* kp = qk + ((size_t)b*NQ + ki) * (2*D_MODEL) + D_MODEL + h*HEAD_DIM;
    float acc = 0.0f;
    #pragma unroll 8
    for (int c = 0; c < HEAD_DIM; ++c) acc += qp[c] * kp[c];
    attn[idx] = acc * 0.14433756729740643f;   // 1/sqrt(48)
}

__global__ __launch_bounds__(128) void k_softmax_row(float* __restrict__ attn, int ncol) {
    const int row = blockIdx.x;
    const int t   = threadIdx.x;
    float* p = attn + (size_t)row * ncol;
    __shared__ float red[128];
    float mx = -1e30f;
    for (int i = t; i < ncol; i += 128) mx = fmaxf(mx, p[i]);
    red[t] = mx; __syncthreads();
    #pragma unroll
    for (int o = 64; o > 0; o >>= 1) { if (t < o) red[t] = fmaxf(red[t], red[t+o]); __syncthreads(); }
    mx = red[0]; __syncthreads();
    float s = 0.0f;
    for (int i = t; i < ncol; i += 128) { float e = expf(p[i] - mx); p[i] = e; s += e; }
    red[t] = s; __syncthreads();
    #pragma unroll
    for (int o = 64; o > 0; o >>= 1) { if (t < o) red[t] += red[t+o]; __syncthreads(); }
    const float inv = 1.0f / red[0];
    for (int i = t; i < ncol; i += 128) p[i] *= inv;
}

__global__ void k_attn_apply(const float* __restrict__ attn, const float* __restrict__ vp,
                             float* __restrict__ out) {
    int idx = blockIdx.x * blockDim.x + threadIdx.x;
    if (idx >= MD_DEC * D_MODEL) return;
    int c384 = idx % D_MODEL;
    int h = c384 / HEAD_DIM;
    int q = (idx / D_MODEL) % NQ;
    int b = idx / (D_MODEL * NQ);
    const float* ap  = attn + (((size_t)b*NHEAD + h)*NQ + q) * NQ;
    const float* vpp = vp + (size_t)b * NQ * D_MODEL + c384;
    float acc = 0.0f;
    for (int k = 0; k < NQ; ++k) acc += ap[k] * vpp[(size_t)k * D_MODEL];
    out[idx] = acc;
}

// ---------------------------------------------------------------------------
// Host side
// ---------------------------------------------------------------------------
extern "C" void kernel_launch(void* const* d_in, const int* in_sizes, int n_in,
                              void* d_out, int out_size, void* d_ws, size_t ws_size,
                              hipStream_t stream) {
    (void)in_sizes; (void)n_in; (void)out_size; (void)ws_size;
    const float* srcs        = (const float*)d_in[0];
    const unsigned char* msk = (const unsigned char*)d_in[1];
    const float* query_embed = (const float*)d_in[2];
    const float* pos_embeds  = (const float*)d_in[3];
    const float* level_embed = (const float*)d_in[4];
    const float* ref_w       = (const float*)d_in[5];
    const float* ref_b       = (const float*)d_in[6];
    const float* const* EIN = (const float* const*)(d_in + 7);    // e_* : 16 entries
    const float* const* DIN = (const float* const*)(d_in + 23);   // d_* : 16 entries
    const float* sa_in_w  = (const float*)d_in[39];
    const float* sa_in_b  = (const float*)d_in[40];
    const float* sa_out_w = (const float*)d_in[41];
    const float* sa_out_b = (const float*)d_in[42];
    const float* n3_g     = (const float*)d_in[43];
    const float* n3_b     = (const float*)d_in[44];

    static const size_t PSTRIDE[16] = {
        96*384, 96, 32*384, 32, 384*384, 384, 384*384, 384,
        384, 384, (size_t)DFFN*384, DFFN, 384*(size_t)DFFN, 384, 384, 384 };

    // ---- workspace bump allocator ----
    char* wsp = (char*)d_ws;
    auto allocB = [&](size_t bytes) -> void* {
        void* p = (void*)wsp;
        wsp += (bytes + 255) & ~(size_t)255;
        return p;
    };
    auto allocF = [&](size_t elems) -> float*    { return (float*)allocB(elems * 4); };
    auto allocH = [&](size_t elems) -> _Float16* { return (_Float16*)allocB(elems * 2); };

    float* bx    = allocF((size_t)MB_ENC * D_MODEL);   // encoder tokens / memory
    float* bpos  = allocF((size_t)MB_ENC * D_MODEL);
    float* bq    = allocF((size_t)MB_ENC * D_MODEL);   // query (x+pos / y+qe)
    float* bval  = allocF((size_t)MB_ENC * D_MODEL);   // MSDA value projection
    float* boff  = allocF((size_t)MB_ENC * 96);
    float* baw   = allocF((size_t)MB_ENC * 32);
    float* bmsda = allocF((size_t)MB_ENC * D_MODEL);   // sampled output / attn out
    float* bt1   = allocF((size_t)MB_ENC * D_MODEL);   // residual branch
    float* bbig  = allocF((size_t)MB_ENC * DFFN);      // FFN hidden / attn matrix
    float* beref = allocF((size_t)MB_ENC * 3);
    float* bvalid= allocF(64);
    float* by    = allocF((size_t)MD_DEC * D_MODEL);
    float* bqe   = allocF((size_t)MD_DEC * D_MODEL);
    float* bdref = allocF((size_t)MD_DEC * 3);
    float* bqk   = allocF((size_t)MD_DEC * 2*D_MODEL);
    float* bvp2  = allocF((size_t)MD_DEC * D_MODEL);
    _Float16* hA = allocH((size_t)MB_ENC * DFFN);      // f16 activations (max size)
    _Float16* hW = allocH((size_t)DFFN * D_MODEL);     // f16 weights (max size)

    auto grid1 = [](int n) { return dim3((n + 255) / 256); };
    auto cvt = [&](const float* src, _Float16* dst, size_t elems) {
        int n4 = (int)(elems / 4);
        k_cvt_f16<<<grid1(n4), 256, 0, stream>>>(src, dst, n4);
    };
    // convert activation + weight, then run the WMMA GEMM (16x64 tiles when possible)
    auto gemm = [&](const float* Af, const float* Wf, const float* bias, float* Y,
                    int M, int N, int K, int relu) {
        cvt(Af, hA, (size_t)M * K);
        cvt(Wf, hW, (size_t)N * K);
        int tilesM = (M + 15) / 16;
        if (N % 64 == 0) {
            int waves = tilesM * (N / 64);
            gemm_wmma_f16<4><<<dim3((waves + 7) / 8), 256, 0, stream>>>(hA, hW, bias, Y, M, N, K, relu);
        } else {
            int waves = tilesM * (N / 32);
            gemm_wmma_f16<2><<<dim3((waves + 7) / 8), 256, 0, stream>>>(hA, hW, bias, Y, M, N, K, relu);
        }
    };
    // variant reusing an already-converted activation (saves re-conversion)
    auto gemmH = [&](const _Float16* Ah, const float* Wf, const float* bias, float* Y,
                     int M, int N, int K, int relu) {
        cvt(Wf, hW, (size_t)N * K);
        int tilesM = (M + 15) / 16;
        if (N % 64 == 0) {
            int waves = tilesM * (N / 64);
            gemm_wmma_f16<4><<<dim3((waves + 7) / 8), 256, 0, stream>>>(Ah, hW, bias, Y, M, N, K, relu);
        } else {
            int waves = tilesM * (N / 32);
            gemm_wmma_f16<2><<<dim3((waves + 7) / 8), 256, 0, stream>>>(Ah, hW, bias, Y, M, N, K, relu);
        }
    };

    // ------------------- prep -------------------
    k_prep_xpos<<<grid1(MB_ENC*D_MODEL), 256, 0, stream>>>(srcs, pos_embeds, level_embed, bx, bpos);
    k_valid<<<1, 32, 0, stream>>>(msk, bvalid);
    k_enc_ref<<<grid1(MB_ENC), 256, 0, stream>>>(bvalid, beref);

    // ------------------- encoder -------------------
    for (int l = 0; l < LENC; ++l) {
        const float* off_w = EIN[0] + (size_t)l * PSTRIDE[0];
        const float* off_b = EIN[1] + (size_t)l * PSTRIDE[1];
        const float* aw_w  = EIN[2] + (size_t)l * PSTRIDE[2];
        const float* aw_b  = EIN[3] + (size_t)l * PSTRIDE[3];
        const float* vp_w  = EIN[4] + (size_t)l * PSTRIDE[4];
        const float* vp_b  = EIN[5] + (size_t)l * PSTRIDE[5];
        const float* op_w  = EIN[6] + (size_t)l * PSTRIDE[6];
        const float* op_b  = EIN[7] + (size_t)l * PSTRIDE[7];
        const float* n1g   = EIN[8] + (size_t)l * 384;
        const float* n1b   = EIN[9] + (size_t)l * 384;
        const float* l1w   = EIN[10] + (size_t)l * PSTRIDE[10];
        const float* l1b   = EIN[11] + (size_t)l * PSTRIDE[11];
        const float* l2w   = EIN[12] + (size_t)l * PSTRIDE[12];
        const float* l2b   = EIN[13] + (size_t)l * PSTRIDE[13];
        const float* n2g   = EIN[14] + (size_t)l * 384;
        const float* n2b   = EIN[15] + (size_t)l * 384;

        k_add2<<<grid1(MB_ENC*D_MODEL), 256, 0, stream>>>(bx, bpos, bq, MB_ENC*D_MODEL);
        gemm(bx, vp_w, vp_b, bval, MB_ENC, D_MODEL, D_MODEL, 0);
        k_mask_zero<<<grid1(MB_ENC*D_MODEL), 256, 0, stream>>>(bval, msk);
        cvt(bq, hA, (size_t)MB_ENC * D_MODEL);                               // q used twice
        gemmH(hA, off_w, off_b, boff, MB_ENC, 96, D_MODEL, 0);
        gemmH(hA, aw_w,  aw_b,  baw,  MB_ENC, 32, D_MODEL, 0);
        k_msda_sample<<<grid1(BS*SVOX*D_MODEL), 256, 0, stream>>>(bval, boff, baw, beref, bmsda, SVOX);
        gemm(bmsda, op_w, op_b, bt1, MB_ENC, D_MODEL, D_MODEL, 0);
        ln_residual<<<MB_ENC, 128, 0, stream>>>(bx, bt1, n1g, n1b, bx);
        gemm(bx,   l1w, l1b, bbig, MB_ENC, DFFN, D_MODEL, 1);
        gemm(bbig, l2w, l2b, bt1,  MB_ENC, D_MODEL, DFFN, 0);
        ln_residual<<<MB_ENC, 128, 0, stream>>>(bx, bt1, n2g, n2b, bx);
    }
    // bx now holds `memory`

    // ------------------- decoder prep -------------------
    k_init_query<<<grid1(MD_DEC*D_MODEL), 256, 0, stream>>>(query_embed, bqe, by);
    k_dec_ref<<<grid1(MD_DEC*3), 256, 0, stream>>>(bqe, ref_w, ref_b, bvalid, bdref);

    // ------------------- decoder -------------------
    for (int l = 0; l < LDEC; ++l) {
        const float* off_w = DIN[0] + (size_t)l * PSTRIDE[0];
        const float* off_b = DIN[1] + (size_t)l * PSTRIDE[1];
        const float* aw_w  = DIN[2] + (size_t)l * PSTRIDE[2];
        const float* aw_b  = DIN[3] + (size_t)l * PSTRIDE[3];
        const float* vp_w  = DIN[4] + (size_t)l * PSTRIDE[4];
        const float* vp_b  = DIN[5] + (size_t)l * PSTRIDE[5];
        const float* op_w  = DIN[6] + (size_t)l * PSTRIDE[6];
        const float* op_b  = DIN[7] + (size_t)l * PSTRIDE[7];
        const float* n1g   = DIN[8] + (size_t)l * 384;
        const float* n1b   = DIN[9] + (size_t)l * 384;
        const float* l1w   = DIN[10] + (size_t)l * PSTRIDE[10];
        const float* l1b   = DIN[11] + (size_t)l * PSTRIDE[11];
        const float* l2w   = DIN[12] + (size_t)l * PSTRIDE[12];
        const float* l2b   = DIN[13] + (size_t)l * PSTRIDE[13];
        const float* n2g   = DIN[14] + (size_t)l * 384;
        const float* n2b   = DIN[15] + (size_t)l * 384;
        const float* saiw  = sa_in_w  + (size_t)l * 3*D_MODEL*D_MODEL;
        const float* saib  = sa_in_b  + (size_t)l * 3*D_MODEL;
        const float* saow  = sa_out_w + (size_t)l * D_MODEL*D_MODEL;
        const float* saob  = sa_out_b + (size_t)l * D_MODEL;
        const float* ln3g  = n3_g + (size_t)l * 384;
        const float* ln3b  = n3_b + (size_t)l * 384;

        // ---- self attention ----
        k_add2<<<grid1(MD_DEC*D_MODEL), 256, 0, stream>>>(by, bqe, bq, MD_DEC*D_MODEL);
        gemm(bq, saiw, saib, bqk, MD_DEC, 2*D_MODEL, D_MODEL, 0);                       // q,k proj
        gemm(by, saiw + (size_t)2*D_MODEL*D_MODEL, saib + 2*D_MODEL, bvp2,
             MD_DEC, D_MODEL, D_MODEL, 0);                                              // v proj
        k_attn_scores<<<grid1(BS*NHEAD*NQ*NQ), 256, 0, stream>>>(bqk, bbig);
        k_softmax_row<<<BS*NHEAD*NQ, 128, 0, stream>>>(bbig, NQ);
        k_attn_apply<<<grid1(MD_DEC*D_MODEL), 256, 0, stream>>>(bbig, bvp2, bmsda);
        gemm(bmsda, saow, saob, bt1, MD_DEC, D_MODEL, D_MODEL, 0);
        ln_residual<<<MD_DEC, 128, 0, stream>>>(by, bt1, n2g, n2b, by);

        // ---- cross MSDA over memory ----
        k_add2<<<grid1(MD_DEC*D_MODEL), 256, 0, stream>>>(by, bqe, bq, MD_DEC*D_MODEL);
        gemm(bx, vp_w, vp_b, bval, MB_ENC, D_MODEL, D_MODEL, 0);
        k_mask_zero<<<grid1(MB_ENC*D_MODEL), 256, 0, stream>>>(bval, msk);
        cvt(bq, hA, (size_t)MD_DEC * D_MODEL);                               // q used twice
        gemmH(hA, off_w, off_b, boff, MD_DEC, 96, D_MODEL, 0);
        gemmH(hA, aw_w,  aw_b,  baw,  MD_DEC, 32, D_MODEL, 0);
        k_msda_sample<<<grid1(BS*NQ*D_MODEL), 256, 0, stream>>>(bval, boff, baw, bdref, bmsda, NQ);
        gemm(bmsda, op_w, op_b, bt1, MD_DEC, D_MODEL, D_MODEL, 0);
        ln_residual<<<MD_DEC, 128, 0, stream>>>(by, bt1, n1g, n1b, by);

        // ---- FFN ----
        gemm(by,   l1w, l1b, bbig, MD_DEC, DFFN, D_MODEL, 1);
        gemm(bbig, l2w, l2b, bt1,  MD_DEC, D_MODEL, DFFN, 0);
        ln_residual<<<MD_DEC, 128, 0, stream>>>(by, bt1, ln3g, ln3b, by);
    }

    k_copy<<<grid1(MD_DEC*D_MODEL), 256, 0, stream>>>(by, (float*)d_out, MD_DEC*D_MODEL);
}
